// Attention_66838281060978
// MI455X (gfx1250) — matrix operand order repaired and checked by reading, not compiled, not run on hardware
//
#include <hip/hip_runtime.h>
#include <hip/hip_bf16.h>
#include <math.h>

typedef unsigned short u16t;
typedef __attribute__((ext_vector_type(16))) __bf16 v16bf;
typedef __attribute__((ext_vector_type(8)))  float  v8f;
typedef __attribute__((ext_vector_type(4)))  int    v4i;

#define EPS_F 1.1920929e-7f

#if defined(__has_builtin)
#if __has_builtin(__builtin_amdgcn_global_load_async_to_lds_b128)
#define USE_ASYNC_LDS 1
#endif
#endif

union Frag16 {
  uint4 u4[2];
  u16t  s[16];
  v16bf v;
};

static __device__ __forceinline__ u16t f2bf(float f) {
  unsigned int u = __builtin_bit_cast(unsigned int, f);
  u += 0x7fffu + ((u >> 16) & 1u);   // round-to-nearest-even
  return (u16t)(u >> 16);
}

static __device__ __forceinline__ v8f wmma_bf16(const Frag16& a, const Frag16& b, v8f c) {
  // (neg_a, A, neg_b, B, c_mod, C, reuse_a, reuse_b)
  return __builtin_amdgcn_wmma_f32_16x16x32_bf16(false, a.v, false, b.v, (short)0, c, false, false);
}

static __device__ __forceinline__ float red_sum32(float v) {
  v += __shfl_xor(v, 16, 32);
  v += __shfl_xor(v, 8, 32);
  v += __shfl_xor(v, 4, 32);
  v += __shfl_xor(v, 2, 32);
  v += __shfl_xor(v, 1, 32);
  return v;
}

#if USE_ASYNC_LDS
typedef __attribute__((address_space(3))) v4i lds_v4i;
typedef __attribute__((address_space(1))) v4i glb_v4i;   // builtin param is non-const
// generic LDS pointer low 32 bits == LDS byte offset (ISA 10.2 aperture rules)
static __device__ __forceinline__ lds_v4i* to_lds(void* p) {
  return (lds_v4i*)(unsigned int)(unsigned long long)p;
}
static __device__ __forceinline__ glb_v4i* to_glb(const void* p) {
  return (glb_v4i*)(unsigned long long)p;
}
template <int N>
static __device__ __forceinline__ void wait_async() {
#if __has_builtin(__builtin_amdgcn_s_wait_asynccnt)
  __builtin_amdgcn_s_wait_asynccnt(N);
#else
  asm volatile("s_wait_asynccnt %0" :: "i"(N) : "memory");
#endif
}
#endif

// ---------------------------------------------------------------- converts
__global__ void k_f32_to_bf16(const float* __restrict__ src, u16t* __restrict__ dst, int n) {
  int i = blockIdx.x * blockDim.x + threadIdx.x;
  int stride = gridDim.x * blockDim.x;
  for (; i < n; i += stride) dst[i] = f2bf(src[i]);
}

// src: K x N row-major fp32 -> dst: N x K row-major bf16 (i.e. transposed)
__global__ void k_transpose_bf16(const float* __restrict__ src, u16t* __restrict__ dst,
                                 int K, int N) {
  int idx = blockIdx.x * blockDim.x + threadIdx.x;
  int total = K * N;
  int stride = gridDim.x * blockDim.x;
  for (; idx < total; idx += stride) {
    int k = idx / N, n = idx - k * N;        // coalesced read
    dst[(size_t)n * K + k] = f2bf(src[idx]);
  }
}

// ---------------------------------------------------------------- bf16 GEMM, B pre-transposed
// C(MxN fp32) = A(MxK bf16 row-major) * Bt(NxK bf16 row-major, i.e. B transposed)
// one wave per 16x64 output tile; all fragments are direct global b128 loads.
__global__ void __launch_bounds__(32) k_gemm_bf16(const u16t* __restrict__ A,
                                                  const u16t* __restrict__ Bt,
                                                  float* __restrict__ C,
                                                  int M, int N, int K) {
  const int lane = threadIdx.x;
  const int n16  = lane & 15;
  const int hi   = lane >> 4;           // 0: lanes 0-15, 1: lanes 16-31
  const int m0   = blockIdx.x * 16;
  const int n0   = blockIdx.y * 64;
  v8f acc[4] = {{}, {}, {}, {}};
  const u16t* arow = A + (size_t)(m0 + n16) * K;
  const u16t* brow = Bt + (size_t)(n0 + n16) * K;
  for (int k0 = 0; k0 < K; k0 += 32) {
    // A fragment: elements {k..k+7, k+16..k+23}, k = k0 + 8*hi
    Frag16 fa;
    const u16t* ap = arow + k0 + hi * 8;
    fa.u4[0] = *(const uint4*)(ap);
    fa.u4[1] = *(const uint4*)(ap + 16);
#pragma unroll
    for (int t = 0; t < 4; ++t) {
      // B fragment: lane = column (n0 + t*16 + n16), elements {k..k+7, k+16..k+23}, k = k0 + 8*hi
      Frag16 fb;
      const u16t* bp = brow + (size_t)t * 16 * K + k0 + hi * 8;
      fb.u4[0] = *(const uint4*)(bp);
      fb.u4[1] = *(const uint4*)(bp + 16);
      acc[t] = wmma_bf16(fa, fb, acc[t]);
    }
  }
#pragma unroll
  for (int t = 0; t < 4; ++t)
#pragma unroll
    for (int r = 0; r < 8; ++r)
      C[(size_t)(m0 + hi * 8 + r) * N + n0 + t * 16 + n16] = acc[t][r];
}

// ---------------------------------------------------------------- qkv post: rmsnorm + rotary -> bf16 Q/K/Vt
__global__ void __launch_bounds__(256) k_qkv_post(const float* __restrict__ qkv,
                                                  const float* __restrict__ rotary,
                                                  const float* __restrict__ wq,
                                                  const float* __restrict__ wk,
                                                  const float* __restrict__ wv,
                                                  u16t* __restrict__ Qb,
                                                  u16t* __restrict__ Kb,
                                                  u16t* __restrict__ Vt) {
  const int lane = threadIdx.x & 31;
  const int t = blockIdx.x * 8 + (threadIdx.x >> 5);   // row 0..4095
  const int b = t >> 11, nn = t & 2047;
  const float* row = qkv + (size_t)t * 1344;
  float pc[4], psn[4];
#pragma unroll
  for (int j = 0; j < 4; ++j) {
    float p = rotary[nn * 128 + lane + 32 * j];
    pc[j] = __cosf(p); psn[j] = __sinf(p);
  }
  // segments 0..7 = q heads, 8 = k
  for (int seg = 0; seg < 9; ++seg) {
    const float* src = row + (seg < 8 ? seg * 128 : 1024);
    const float* w = (seg < 8) ? wq : wk;
    float x[4], ss = 0.f;
#pragma unroll
    for (int j = 0; j < 4; ++j) { x[j] = src[lane + 32 * j]; ss += x[j] * x[j]; }
    ss = red_sum32(ss);
    float rms = rsqrtf(ss * (1.f / 128.f) + EPS_F);
    float sc = (seg < 8) ? (rms * 0.125f) : rms;   // q scaled by 64^-0.5
    float nv[4];
#pragma unroll
    for (int j = 0; j < 4; ++j) nv[j] = x[j] * sc * w[lane + 32 * j];
    // rotate_half pairs: lane holds dims l, l+32, l+64, l+96
    float r0 = nv[0] * pc[0] - nv[2] * psn[0];
    float r1 = nv[1] * pc[1] - nv[3] * psn[1];
    float r2 = nv[2] * pc[2] + nv[0] * psn[2];
    float r3 = nv[3] * pc[3] + nv[1] * psn[3];
    u16t* dst = (seg < 8)
        ? Qb + (((size_t)(b * 8 + seg) * 2048) + nn) * 128
        : Kb + ((size_t)b * 2048 + nn) * 128;
    dst[lane]      = f2bf(r0);
    dst[lane + 32] = f2bf(r1);
    dst[lane + 64] = f2bf(r2);
    dst[lane + 96] = f2bf(r3);
  }
  // v: 192 dims, stored transposed per 32-key block: Vt[b][nn>>5][dim][nn&31]
  {
    const float* src = row + 1152;
    float x[6], ss = 0.f;
#pragma unroll
    for (int j = 0; j < 6; ++j) { x[j] = src[lane + 32 * j]; ss += x[j] * x[j]; }
    ss = red_sum32(ss);
    float rms = rsqrtf(ss * (1.f / 192.f) + EPS_F);
    u16t* base = Vt + ((size_t)(b * 64 + (nn >> 5)) * 192) * 32 + (nn & 31);
#pragma unroll
    for (int j = 0; j < 6; ++j)
      base[(size_t)(lane + 32 * j) * 32] = f2bf(x[j] * rms * wv[lane + 32 * j]);
  }
}

// ---------------------------------------------------------------- pairwise bias: rmsnorm -> gelu -> (128x8) GEMV
__global__ void __launch_bounds__(256) k_bias(const float* __restrict__ pw,
                                              const float* __restrict__ wnorm,
                                              const float* __restrict__ Wb,
                                              float* __restrict__ bias) {
  const int lane = threadIdx.x & 31;
  const long t = (long)blockIdx.x * 8 + (threadIdx.x >> 5);   // (b,i,j) pixel
  const int b = (int)(t >> 18);
  const int rem = (int)(t & 262143);
  const int i = rem >> 9, j = rem & 511;
  const float* row = pw + ((size_t)(b * 512 + i) * 512 + j) * 128;
  float x[4], ss = 0.f;
#pragma unroll
  for (int c = 0; c < 4; ++c) { x[c] = row[lane + 32 * c]; ss += x[c] * x[c]; }
  ss = red_sum32(ss);
  float rms = rsqrtf(ss * (1.f / 128.f) + EPS_F);
  float acc[8];
#pragma unroll
  for (int h = 0; h < 8; ++h) acc[h] = 0.f;
#pragma unroll
  for (int c = 0; c < 4; ++c) {
    int d = lane + 32 * c;
    float g = x[c] * rms * wnorm[d];
    g = 0.5f * g * (1.f + erff(g * 0.70710678118f));   // exact GELU
#pragma unroll
    for (int h = 0; h < 8; ++h) acc[h] += g * Wb[d * 8 + h];
  }
#pragma unroll
  for (int h = 0; h < 8; ++h) acc[h] = red_sum32(acc[h]);
  if (lane < 8)
    bias[((size_t)(b * 8 + lane) * 512 + i) * 512 + j] = acc[lane];
}

// ---------------------------------------------------------------- flash attention
// grid (b*H=16, n/64=32), block 128 (4 waves); wave handles 16 queries.
// Double-buffered async K/V tile pipeline: issue tile i+1, wait asynccnt<=10
// (tile i resident, tile i+1 in flight), compute tile i.
__global__ void __launch_bounds__(128) k_flash(const u16t* __restrict__ Qb,
                                               const u16t* __restrict__ Kb,
                                               const u16t* __restrict__ Vt,
                                               const float* __restrict__ bias,
                                               u16t* __restrict__ Ob) {
  __shared__ u16t sK[2][32 * 128];    // [buf][key][dim]
  __shared__ u16t sVt[2][192 * 32];   // [buf][dim][key]
  __shared__ u16t sP[4][16 * 32];
  const int tid = threadIdx.x;
  const int lane = tid & 31, wave = tid >> 5;
  const int bh = blockIdx.x;
  const int b = bh >> 3, h = bh & 7;
  const int q0 = blockIdx.y * 64 + wave * 16;
  const int n16 = lane & 15;
  const int hi = lane >> 4;

  // issue one 32-key tile fill (10 async/plain b128 transfers per thread)
  auto issue_tile = [&](int kb, int buf) {
    const uint4* gk = (const uint4*)(Kb + ((size_t)b * 2048 + kb) * 128);
    const uint4* gv = (const uint4*)(Vt + ((size_t)(b * 64 + (kb >> 5)) * 192) * 32);
#if USE_ASYNC_LDS
    for (int idx = tid; idx < 512; idx += 128)
      __builtin_amdgcn_global_load_async_to_lds_b128(to_glb(gk + idx), to_lds((uint4*)sK[buf] + idx), 0, 0);
    for (int idx = tid; idx < 768; idx += 128)
      __builtin_amdgcn_global_load_async_to_lds_b128(to_glb(gv + idx), to_lds((uint4*)sVt[buf] + idx), 0, 0);
#else
    uint4* dk = (uint4*)sK[buf];
    for (int idx = tid; idx < 512; idx += 128) dk[idx] = gk[idx];
    uint4* dv = (uint4*)sVt[buf];
    for (int idx = tid; idx < 768; idx += 128) dv[idx] = gv[idx];
#endif
  };

  // Q A-fragments (K=128 split into 4 chunks of 32)
  Frag16 qa[4];
  {
    const u16t* qrow = Qb + (((size_t)(b * 8 + h) * 2048) + q0 + n16) * 128;
#pragma unroll
    for (int ks = 0; ks < 4; ++ks) {
      const u16t* p = qrow + ks * 32 + hi * 8;
      qa[ks].u4[0] = *(const uint4*)(p);
      qa[ks].u4[1] = *(const uint4*)(p + 16);
    }
  }
  v8f o[12];
#pragma unroll
  for (int nb = 0; nb < 12; ++nb) o[nb] = v8f{};
  float mrow[8], lrow[8];
#pragma unroll
  for (int r = 0; r < 8; ++r) { mrow[r] = -1e30f; lrow[r] = 0.f; }

  const float* biasbh = bias + (size_t)(b * 8 + h) * 512 * 512;

  issue_tile(0, 0);
  for (int it = 0; it < 64; ++it) {
    const int kb = it * 32;
    const int buf = it & 1;
    const bool more = (it + 1) < 64;
    if (more) issue_tile(kb + 32, buf ^ 1);   // prior-iteration end barrier closed WAR window
#if USE_ASYNC_LDS
    if (more) wait_async<10>();               // tile `it` resident; tile `it+1` in flight
    else      wait_async<0>();
#endif
    __syncthreads();

    // S = Q K^T (16 queries x 32 keys)
    const u16t* cK = sK[buf];
    const u16t* cV = sVt[buf];
    v8f s[2];
#pragma unroll
    for (int nh = 0; nh < 2; ++nh) {
      v8f acc = {};
#pragma unroll
      for (int ks = 0; ks < 4; ++ks) {
        // B[k][n] = Ktile[key n][dim ks*32 + hi*8 + {0..7, 16..23}]
        Frag16 fb;
        const u16t* p = cK + (nh * 16 + n16) * 128 + ks * 32 + hi * 8;
        fb.u4[0] = *(const uint4*)(p);
        fb.u4[1] = *(const uint4*)(p + 16);
        acc = wmma_bf16(qa[ks], fb, acc);
      }
      s[nh] = acc;
    }

    // bias + softclamp + online softmax (C layout: lane=col, rows hi*8+r)
    const int rb = hi * 8;
    v8f scalev;
    float pr0[8], pr1[8];
#pragma unroll
    for (int r = 0; r < 8; ++r) {
      int irow = (q0 + rb + r) >> 2;
      float s0 = s[0][r] + biasbh[(size_t)irow * 512 + ((kb + n16) >> 2)];
      float s1 = s[1][r] + biasbh[(size_t)irow * 512 + ((kb + 16 + n16) >> 2)];
      s0 = 5.f * tanhf(s0 * 0.2f);
      s1 = 5.f * tanhf(s1 * 0.2f);
      float mx = fmaxf(s0, s1);
      mx = fmaxf(mx, __shfl_xor(mx, 1, 32));
      mx = fmaxf(mx, __shfl_xor(mx, 2, 32));
      mx = fmaxf(mx, __shfl_xor(mx, 4, 32));
      mx = fmaxf(mx, __shfl_xor(mx, 8, 32));
      float nm = fmaxf(mrow[r], mx);
      float sc = __expf(mrow[r] - nm);
      mrow[r] = nm;
      float p0 = __expf(s0 - nm), p1 = __expf(s1 - nm);
      float psum = p0 + p1;
      psum += __shfl_xor(psum, 1, 32);
      psum += __shfl_xor(psum, 2, 32);
      psum += __shfl_xor(psum, 4, 32);
      psum += __shfl_xor(psum, 8, 32);
      lrow[r] = lrow[r] * sc + psum;
      scalev[r] = sc;
      pr0[r] = p0; pr1[r] = p1;
    }
#pragma unroll
    for (int nb = 0; nb < 12; ++nb) o[nb] *= scalev;

    // C-layout -> A-layout for P via LDS round trip
    u16t* myP = sP[wave];
#pragma unroll
    for (int r = 0; r < 8; ++r) {
      myP[(rb + r) * 32 + n16]      = f2bf(pr0[r]);
      myP[(rb + r) * 32 + 16 + n16] = f2bf(pr1[r]);
    }
    __syncthreads();
    Frag16 pa;
    {
      const u16t* p = myP + n16 * 32 + hi * 8;
      pa.u4[0] = *(const uint4*)(p);
      pa.u4[1] = *(const uint4*)(p + 16);
    }
    // O += P V  (16x32 @ 32x192); V-fragment contiguous thanks to transposed tile
#pragma unroll
    for (int nb = 0; nb < 12; ++nb) {
      Frag16 fv;
      const u16t* p = cV + (nb * 16 + n16) * 32 + hi * 8;
      fv.u4[0] = *(const uint4*)(p);
      fv.u4[1] = *(const uint4*)(p + 16);
      o[nb] = wmma_bf16(pa, fv, o[nb]);
    }
    __syncthreads();   // everyone done reading buf before it is refilled next iteration
  }

  // epilogue: normalize and emit bf16 attn-out, layout [b][n][h*192+d]
#pragma unroll
  for (int r = 0; r < 8; ++r) {
    float inv = 1.f / lrow[r];
    int irow = q0 + hi * 8 + r;
    u16t* orow = Ob + ((size_t)b * 2048 + irow) * 1536 + h * 192;
#pragma unroll
    for (int nb = 0; nb < 12; ++nb)
      orow[nb * 16 + n16] = f2bf(o[nb][r] * inv);
  }
}

// ---------------------------------------------------------------- host launch
extern "C" void kernel_launch(void* const* d_in, const int* in_sizes, int n_in,
                              void* d_out, int out_size, void* d_ws, size_t ws_size,
                              hipStream_t stream) {
  (void)in_sizes; (void)n_in; (void)out_size; (void)ws_size;
  const float* x      = (const float*)d_in[0];
  const float* pw     = (const float*)d_in[1];
  const float* rotary = (const float*)d_in[2];
  const float* W_qkv  = (const float*)d_in[3];
  const float* W_out  = (const float*)d_in[4];
  const float* w_q    = (const float*)d_in[5];
  const float* w_k    = (const float*)d_in[6];
  const float* w_v    = (const float*)d_in[7];
  const float* w_pw   = (const float*)d_in[8];
  const float* W_bias = (const float*)d_in[9];
  float* out = (float*)d_out;

  char* ws = (char*)d_ws;
  size_t off = 0;
  auto alloc = [&](size_t bytes) {
    void* p = ws + off;
    off += (bytes + 255) & ~(size_t)255;
    return p;
  };
  u16t*  xb     = (u16t*)alloc((size_t)4096 * 512 * 2);
  u16t*  wqkvt  = (u16t*)alloc((size_t)1344 * 512 * 2);   // W_qkv^T
  u16t*  woutt  = (u16t*)alloc((size_t)512 * 1536 * 2);   // W_out^T
  float* qkv    = (float*)alloc((size_t)4096 * 1344 * 4);
  u16t*  Qb     = (u16t*)alloc((size_t)2 * 8 * 2048 * 128 * 2);
  u16t*  Kb     = (u16t*)alloc((size_t)2 * 2048 * 128 * 2);
  u16t*  Vt     = (u16t*)alloc((size_t)2 * 64 * 192 * 32 * 2);
  float* biasbf = (float*)alloc((size_t)2 * 8 * 512 * 512 * 4);
  u16t*  attnb  = (u16t*)alloc((size_t)4096 * 1536 * 2);

  k_f32_to_bf16<<<2048, 256, 0, stream>>>(x, xb, 4096 * 512);
  k_transpose_bf16<<<1024, 256, 0, stream>>>(W_qkv, wqkvt, 512, 1344);
  k_transpose_bf16<<<1024, 256, 0, stream>>>(W_out, woutt, 1536, 512);

  // qkv = x @ W_qkv : (4096 x 512) @ (512 x 1344), 16x64 tiles
  k_gemm_bf16<<<dim3(256, 21), 32, 0, stream>>>(xb, wqkvt, qkv, 4096, 1344, 512);

  k_qkv_post<<<512, 256, 0, stream>>>(qkv, rotary, w_q, w_k, w_v, Qb, Kb, Vt);

  k_bias<<<65536, 256, 0, stream>>>(pw, w_pw, W_bias, biasbf);

  k_flash<<<dim3(16, 32), 128, 0, stream>>>(Qb, Kb, Vt, biasbf, attnb);

  // out = attn_out @ W_out : (4096 x 1536) @ (1536 x 512)
  k_gemm_bf16<<<dim3(256, 8), 32, 0, stream>>>(attnb, woutt, out, 4096, 512, 1536);
}
